// RNNModelScratch_10514079940714
// MI455X (gfx1250) — compile-verified, matrix operand order checked by reading, not run on hardware
//
#include <hip/hip_runtime.h>

typedef __attribute__((ext_vector_type(16))) _Float16 v16h;
typedef __attribute__((ext_vector_type(8)))  _Float16 v8h;
typedef __attribute__((ext_vector_type(8)))  float    v8f;

#define VSZ 1024
#define NH  256
#define BB  128
#define TT  512
#define NWG 16
#define TBV (TT*BB*VSZ)

// LDS layout (halves): sW1[4][16][264] | sW2[4][16][520] | sWo[64][264]
#define SW1_OFF 0
#define SW2_OFF 16896
#define SWO_OFF (16896+33280)
#define SMEM_HALVES (16896+33280+16896)   // 67072 halves = 134144 bytes

#define WMMA(a,b,c) __builtin_amdgcn_wmma_f32_16x16x32_f16(false,(a),false,(b),(short)0,(c),false,false)

__device__ __forceinline__ float fsig(float x){ return 1.f/(1.f+__expf(-x)); }
__device__ __forceinline__ float ftanh(float x){ return 2.f/(1.f+__expf(-2.f*x)) - 1.f; }

// Load a 16x32 f16 WMMA operand fragment for this lane:
// elements 0..7 = p[0..7], elements 8..15 = p[16..23]  (two b128 loads)
__device__ __forceinline__ v16h ldAB(const _Float16* p){
  v8h lo = *(const v8h*)p;
  v8h hi = *(const v8h*)(p+16);
  v16h r;
#pragma unroll
  for(int i=0;i<8;++i){ r[i]=lo[i]; r[8+i]=hi[i]; }
  return r;
}

// -------- one-time prep: f16 transposed weights + H state init + barrier init
__global__ void prep(const float* __restrict__ W1f,const float* __restrict__ W1i,
                     const float* __restrict__ W1o,const float* __restrict__ W1c,
                     const float* __restrict__ W2f,const float* __restrict__ W2i,
                     const float* __restrict__ W2o,const float* __restrict__ W2c,
                     const float* __restrict__ Wout,
                     const float* __restrict__ H1in,const float* __restrict__ H2in,
                     _Float16* __restrict__ U1T,_Float16* __restrict__ W2T,
                     _Float16* __restrict__ WoutT,
                     _Float16* __restrict__ H1b,_Float16* __restrict__ H2b,
                     unsigned* __restrict__ bar){
  const float* W1s[4]={W1f,W1i,W1o,W1c};
  const float* W2s[4]={W2f,W2i,W2o,W2c};
  for(long long i=(long long)blockIdx.x*256+threadIdx.x; i<1114114LL; i+=(long long)gridDim.x*256){
    long long j=i;
    if(j<262144){ // U1T[g][n][k] = W1g[(V+k)*NH + n]   (hidden part, transposed)
      int g=(int)(j>>16); int r=(int)(j&65535); int n=r>>8; int k=r&255;
      U1T[j]=(_Float16)W1s[g][(VSZ+k)*NH+n]; continue; }
    j-=262144;
    if(j<524288){ // W2T[g][n][k] = W2g[k*NH + n], k in [0,512)
      int g=(int)(j>>17); int r=(int)(j&131071); int n=r>>9; int k=r&511;
      W2T[j]=(_Float16)W2s[g][k*NH+n]; continue; }
    j-=524288;
    if(j<262144){ // WoutT[n][k] = Wout[k*V + n]
      int n=(int)(j>>8); int k=(int)(j&255);
      WoutT[j]=(_Float16)Wout[k*VSZ+n]; continue; }
    j-=262144;
    if(j<32768){ H1b[j]=(_Float16)H1in[j]; continue; }
    j-=32768;
    if(j<32768){ H2b[j]=(_Float16)H2in[j]; continue; }
    j-=32768;
    if(j<2) bar[j]=0u;
  }
}

// -------- device-scope barrier across the 16 persistent workgroups
__device__ __forceinline__ void gridbar(unsigned* bar){
  __syncthreads();
  __threadfence();
  if(threadIdx.x==0){
    unsigned g = __hip_atomic_load(&bar[1], __ATOMIC_RELAXED, __HIP_MEMORY_SCOPE_AGENT);
    unsigned prev = __hip_atomic_fetch_add(&bar[0], 1u, __ATOMIC_ACQ_REL, __HIP_MEMORY_SCOPE_AGENT);
    if(prev==NWG-1){
      __hip_atomic_store(&bar[0], 0u, __ATOMIC_RELAXED, __HIP_MEMORY_SCOPE_AGENT);
      __hip_atomic_fetch_add(&bar[1], 1u, __ATOMIC_RELEASE, __HIP_MEMORY_SCOPE_AGENT);
    }else{
      while(__hip_atomic_load(&bar[1], __ATOMIC_ACQUIRE, __HIP_MEMORY_SCOPE_AGENT)==g){
        __builtin_amdgcn_s_sleep(2);
      }
    }
  }
  __syncthreads();
}

// -------- persistent 2-layer LSTM scan. 16 WGs x 256 threads (8 waves).
// WG wg owns cols [wg*16, wg*16+16) of NH (gates) and [wg*64, wg*64+64) of V (output).
// Wave wv owns rows [wv*16, wv*16+16) of B.
__global__ __launch_bounds__(256) void lstm_persist(
    const int* __restrict__ X,
    const float* __restrict__ C1in, const float* __restrict__ C2in,
    const float* __restrict__ W1f,const float* __restrict__ W1i,
    const float* __restrict__ W1o,const float* __restrict__ W1c,
    const float* __restrict__ b1f,const float* __restrict__ b1i,
    const float* __restrict__ b1o,const float* __restrict__ b1c,
    const float* __restrict__ b2f,const float* __restrict__ b2i,
    const float* __restrict__ b2o,const float* __restrict__ b2c,
    const float* __restrict__ bout,
    const _Float16* __restrict__ U1T, const _Float16* __restrict__ W2T,
    const _Float16* __restrict__ WoutT,
    _Float16* __restrict__ H1b, _Float16* __restrict__ H2b,
    unsigned* __restrict__ bar, float* __restrict__ out)
{
  extern __shared__ _Float16 smem[];
  _Float16* sW1 = smem + SW1_OFF;   // [4][16][264]  layer-1 hidden weight slice
  _Float16* sW2 = smem + SW2_OFF;   // [4][16][520]  layer-2 weight slice
  _Float16* sWo = smem + SWO_OFF;   // [64][264]     output weight slice

  const int tid=threadIdx.x, wg=blockIdx.x;
  const int lane=tid&31, wv=tid>>5;
  const int ln=lane&15, hs=lane>>4;      // half-wave select (K phase)
  const int cs=wg*16, colD=cs+ln;        // gate output column of this lane
  const int row0=wv*16, arow=row0+ln;    // A-operand row of this lane

  // stage all per-WG weight slices once (constant for all 512 steps).
  // Keeping B operands in LDS (not registers) stops the compiler from
  // pinning ~500 VGPRs of weights and spilling to scratch.
  for(int e=tid;e<16384;e+=256){                 // layer-1: 4*16*256
    int g=e>>12, n=(e>>8)&15, k=e&255;
    sW1[(g*16+n)*264+k]=U1T[((g*256+cs+n)<<8)+k];
  }
  for(int e=tid;e<32768;e+=256){                 // layer-2: 4*16*512
    int g=e>>13, n=(e>>9)&15, k=e&511;
    sW2[(g*16+n)*520+k]=W2T[((g*256+cs+n)<<9)+k];
  }
  for(int e=tid;e<16384;e+=256){                 // out: 64*256
    int n=e>>8, k=e&255;
    sWo[n*264+k]=WoutT[((wg*64+n)<<8)+k];
  }
  __syncthreads();

  const float* b1s[4]={b1f,b1i,b1o,b1c};
  const float* W1s[4]={W1f,W1i,W1o,W1c};
  const float* b2s[4]={b2f,b2i,b2o,b2c};

  // cell states live in registers, D-fragment layout
  float c1r[8], c2r[8];
#pragma unroll
  for(int v=0;v<8;++v){
    int rowD=row0+v+hs*8;
    c1r[v]=C1in[rowD*NH+colD];
    c2r[v]=C2in[rowD*NH+colD];
  }

  float* outH1=out+(size_t)TBV;
  float* outC1=outH1+32768;
  float* outH2=outC1+32768;
  float* outC2=outH2+32768;

  for(int t=0;t<TT;++t){
    const int p=t&1;
    const _Float16* H1p=H1b+p*32768;
    _Float16*       H1n=H1b+(1-p)*32768;
    const _Float16* H2p=H2b+p*32768;
    _Float16*       H2n=H2b+(1-p)*32768;

    // ---------- phase 1: layer-1 gates = H1 @ U1 + b1 + W1[x,:] gather ----------
    v8f acc[4];
#pragma unroll
    for(int g=0;g<4;++g)
#pragma unroll
      for(int i=0;i<8;++i) acc[g][i]=0.f;

#pragma unroll
    for(int kc=0;kc<8;++kc){
      v16h a = ldAB(H1p + arow*NH + hs*8 + kc*32);   // A reused across 4 gates
#pragma unroll
      for(int g=0;g<4;++g){
        v16h b = ldAB(sW1 + (g*16+ln)*264 + hs*8 + kc*32);
        acc[g]=WMMA(a,b,acc[g]);
      }
    }
#pragma unroll
    for(int v=0;v<8;++v){
      int rowD=row0+v+hs*8;
      int xi=X[rowD*TT+t];                           // one-hot -> row gather
      float pf=acc[0][v]+b1s[0][colD]+W1s[0][xi*NH+colD];
      float pi=acc[1][v]+b1s[1][colD]+W1s[1][xi*NH+colD];
      float po=acc[2][v]+b1s[2][colD]+W1s[2][xi*NH+colD];
      float pc=acc[3][v]+b1s[3][colD]+W1s[3][xi*NH+colD];
      float f=fsig(pf), i_=fsig(pi), o_=fsig(po), ct=ftanh(pc);
      float c=f*c1r[v]+i_*ct; c1r[v]=c;
      float h=o_*ftanh(c);
      H1n[rowD*NH+colD]=(_Float16)h;
      if(t==TT-1){ outH1[rowD*NH+colD]=h; outC1[rowD*NH+colD]=c; }
    }
    gridbar(bar);

    // ---------- phase 2: layer-2 gates = concat(h1_new,h2_prev) @ W2 + b2 ----------
#pragma unroll
    for(int g=0;g<4;++g)
#pragma unroll
      for(int i=0;i<8;++i) acc[g][i]=0.f;

#pragma unroll
    for(int kc=0;kc<16;++kc){
      int off=hs*8+kc*32;
      const _Float16* ap=(off<256)?(H1n+arow*NH+off):(H2p+arow*NH+(off-256));
      v16h a=ldAB(ap);
#pragma unroll
      for(int g=0;g<4;++g){
        v16h b=ldAB(sW2 + (g*16+ln)*520 + off);
        acc[g]=WMMA(a,b,acc[g]);
      }
    }
#pragma unroll
    for(int v=0;v<8;++v){
      int rowD=row0+v+hs*8;
      float pf=acc[0][v]+b2s[0][colD];
      float pi=acc[1][v]+b2s[1][colD];
      float po=acc[2][v]+b2s[2][colD];
      float pc=acc[3][v]+b2s[3][colD];
      float f=fsig(pf), i_=fsig(pi), o_=fsig(po), ct=ftanh(pc);
      float c=f*c2r[v]+i_*ct; c2r[v]=c;
      float h=o_*ftanh(c);
      H2n[rowD*NH+colD]=(_Float16)h;
      if(t==TT-1){ outH2[rowD*NH+colD]=h; outC2[rowD*NH+colD]=c; }
    }
    gridbar(bar);

    // ---------- phase 3: Y[t] = h2_new @ Wout + bout (64 cols per WG) ----------
    v8f ya[4];
#pragma unroll
    for(int nt=0;nt<4;++nt)
#pragma unroll
      for(int i=0;i<8;++i) ya[nt][i]=0.f;

#pragma unroll
    for(int kc=0;kc<8;++kc){
      v16h a=ldAB(H2n + arow*NH + hs*8 + kc*32);
#pragma unroll
      for(int nt=0;nt<4;++nt){
        v16h b=ldAB(sWo + (nt*16+ln)*264 + hs*8 + kc*32);
        ya[nt]=WMMA(a,b,ya[nt]);
      }
    }
#pragma unroll
    for(int nt=0;nt<4;++nt){
      int col=wg*64+nt*16+ln;
      float bo=bout[col];
#pragma unroll
      for(int v=0;v<8;++v){
        int rowD=row0+v+hs*8;
        out[(size_t)(t*BB+rowD)*VSZ+col]=ya[nt][v]+bo;
      }
    }
  }
}

extern "C" void kernel_launch(void* const* d_in, const int* in_sizes, int n_in,
                              void* d_out, int out_size, void* d_ws, size_t ws_size,
                              hipStream_t stream) {
  (void)in_sizes;(void)n_in;(void)out_size;(void)ws_size;
  const int*   X   =(const int*)  d_in[0];
  const float* H1in=(const float*)d_in[1];
  const float* C1in=(const float*)d_in[2];
  const float* H2in=(const float*)d_in[3];
  const float* C2in=(const float*)d_in[4];
  const float* W1f=(const float*)d_in[5];  const float* b1f=(const float*)d_in[6];
  const float* W1i=(const float*)d_in[7];  const float* b1i=(const float*)d_in[8];
  const float* W1o=(const float*)d_in[9];  const float* b1o=(const float*)d_in[10];
  const float* W1c=(const float*)d_in[11]; const float* b1c=(const float*)d_in[12];
  const float* W2f=(const float*)d_in[13]; const float* b2f=(const float*)d_in[14];
  const float* W2i=(const float*)d_in[15]; const float* b2i=(const float*)d_in[16];
  const float* W2o=(const float*)d_in[17]; const float* b2o=(const float*)d_in[18];
  const float* W2c=(const float*)d_in[19]; const float* b2c=(const float*)d_in[20];
  const float* Wout=(const float*)d_in[21];const float* bout=(const float*)d_in[22];

  unsigned* bar  =(unsigned*)d_ws;
  _Float16* U1T  =(_Float16*)((char*)d_ws+64);
  _Float16* W2T  =U1T+262144;
  _Float16* WoutT=W2T+524288;
  _Float16* H1b  =WoutT+262144;   // 2 x [128,256] double buffer
  _Float16* H2b  =H1b+65536;

  prep<<<4353,256,0,stream>>>(W1f,W1i,W1o,W1c,W2f,W2i,W2o,W2c,Wout,
                              H1in,H2in,U1T,W2T,WoutT,H1b,H2b,bar);
  lstm_persist<<<NWG,256,(unsigned)(SMEM_HALVES*sizeof(_Float16)),stream>>>(
                                     X,C1in,C2in,
                                     W1f,W1i,W1o,W1c,
                                     b1f,b1i,b1o,b1c,
                                     b2f,b2i,b2o,b2c,
                                     bout,U1T,W2T,WoutT,
                                     H1b,H2b,bar,(float*)d_out);
}
